// Decoder_72524817760961
// MI455X (gfx1250) — matrix operand order repaired and checked by reading, not compiled, not run on hardware
//
#include <hip/hip_runtime.h>

// ---------------- problem constants ----------------
#define B_  64
#define P_  196
#define D_  2048
#define E_  512
#define H_  512
#define A_  512
#define V_  32000
#define S_  22
#define T_  20   // S - 2

typedef __attribute__((ext_vector_type(16))) __bf16 v16bf;
typedef __attribute__((ext_vector_type(8)))  float  v8f;

__device__ __forceinline__ unsigned short f2bf(float f) {
    unsigned int u = __float_as_uint(f);
    u += 0x7fffu + ((u >> 16) & 1u);           // round-to-nearest-even
    return (unsigned short)(u >> 16);
}
__device__ __forceinline__ float bf2f(unsigned short h) {
    return __uint_as_float(((unsigned int)h) << 16);
}
__device__ __forceinline__ float sigf(float x) { return 1.f / (1.f + __expf(-x)); }

// ---------------- sort / gather setup ----------------
__global__ void k_sort(const int* __restrict__ lengths, const int* __restrict__ captions,
                       int* sort_ind, int* lens_s, int* dec_len, int* caps_s,
                       float* out_caps, float* out_lens, float* out_sortind) {
    __shared__ int sh[B_];
    int i = threadIdx.x;
    if (i < B_) {
        int li = lengths[i], rank = 0;
        for (int j = 0; j < B_; ++j) {
            int lj = lengths[j];
            if (lj > li || (lj == li && j < i)) rank++;   // stable descending
        }
        sh[rank] = i;
    }
    __syncthreads();
    if (i < B_) {
        int src = sh[i];
        sort_ind[i] = src;
        int L = lengths[src];
        lens_s[i] = L; dec_len[i] = L - 1;
        out_lens[i] = (float)L;
        out_sortind[i] = (float)src;
        for (int s = 0; s < S_; ++s) {
            int cv = captions[s * B_ + src];
            caps_s[i * S_ + s] = cv;
            out_caps[i * S_ + s] = (float)cv;
        }
    }
}

__global__ void k_gather_enc(const float* __restrict__ feat, const int* __restrict__ sort_ind,
                             unsigned short* __restrict__ enc_bf) {
    size_t tid = (size_t)blockIdx.x * blockDim.x + threadIdx.x;
    const size_t PD = (size_t)P_ * D_;
    size_t b = tid / PD, rem = tid - b * PD;
    enc_bf[tid] = f2bf(feat[(size_t)sort_ind[b] * PD + rem]);
}

__global__ void k_mean(const unsigned short* __restrict__ enc_bf, unsigned short* __restrict__ mean_bf) {
    int tid = blockIdx.x * blockDim.x + threadIdx.x;   // B*D
    int b = tid / D_, d = tid - b * D_;
    float s = 0.f;
    const unsigned short* p = enc_bf + (size_t)b * P_ * D_ + d;
    for (int pp = 0; pp < P_; ++pp) s += bf2f(p[(size_t)pp * D_]);
    mean_bf[tid] = f2bf(s * (1.f / (float)P_));
}

// ---------------- weight conversion ----------------
__global__ void k_conv(const float* __restrict__ src, unsigned short* __restrict__ dst, long long n) {
    long long tid = (long long)blockIdx.x * blockDim.x + threadIdx.x;
    if (tid < n) dst[tid] = f2bf(src[tid]);
}
__global__ void k_convT(const float* __restrict__ W, unsigned short* __restrict__ Wt, int K, int N) {
    long long tid = (long long)blockIdx.x * blockDim.x + threadIdx.x;
    if (tid < (long long)K * N) {
        int k = (int)(tid / N), n = (int)(tid - (long long)k * N);
        Wt[(size_t)n * K + k] = f2bf(W[tid]);
    }
}
__global__ void k_bsum(const float* a, const float* b, float* o, int n) {
    int i = blockIdx.x * blockDim.x + threadIdx.x;
    if (i < n) o[i] = a[i] + b[i];
}

// ---------------- WMMA GEMM: C[M,N] = A[M,K](bf16) x Bt[N,K](bf16)^T (+bias, epilogue) ----
// Requirements: K % 64 == 0, lda/ldb % 8 == 0 (16-byte aligned rows).
// Branch-free, double-buffered inner loop: out-of-range rows/cols are CLAMPED
// for loads (garbage products land only in rows/cols the epilogue discards,
// since WMMA A-row m / B-col n feed only C[m,*] / C[*,n]).
// flags: 1=accumulate into C, 2=sigmoid, 4=mask rows by dec_len>t (else write 0)
struct Frag { union { uint4 u[2]; v16bf v; }; };

__global__ void __launch_bounds__(32)
k_gemm(const unsigned short* __restrict__ A, int lda, long long sA,
       const unsigned short* __restrict__ Bt, int ldb,
       float* __restrict__ C, int ldc, long long sC,
       const float* __restrict__ bias,
       int M, int N, int K, int flags,
       const int* __restrict__ dec_len, int t) {
    const int lane = threadIdx.x & 31;
    const int half = lane >> 4, sub = lane & 15;
    const int m0 = blockIdx.x * 16;
    const int n0 = blockIdx.y * 64;
    A += (long long)blockIdx.z * sA;
    C += (long long)blockIdx.z * sC;

    v8f acc[4] = {v8f{}, v8f{}, v8f{}, v8f{}};

    // clamped per-lane base pointers (hoisted out of the K loop)
    const int arow = min(m0 + sub, M - 1);
    // A frag (ISA 16-bit A 16x32): lanes 0-15 -> K[0..7]+[16..23], lanes 16-31 -> K[8..15]+[24..31]
    const unsigned short* Ap = A + (size_t)arow * lda + half * 8;
    const unsigned short* Bp[4];
#pragma unroll
    for (int j = 0; j < 4; ++j) {
        int col = min(n0 + j * 16 + sub, N - 1);
        // B frag: lane = column, lanes 0-15 -> K[0..7]+[16..23], lanes 16-31 -> K[8..15]+[24..31]
        Bp[j] = Bt + (size_t)col * ldb + half * 8;
    }

#define LOADA(f, k) { (f).u[0] = *reinterpret_cast<const uint4*>(Ap + (k));        \
                      (f).u[1] = *reinterpret_cast<const uint4*>(Ap + (k) + 16); }
#define LOADB(f, j, k) { (f).u[0] = *reinterpret_cast<const uint4*>(Bp[j] + (k));     \
                         (f).u[1] = *reinterpret_cast<const uint4*>(Bp[j] + (k) + 16); }

    Frag a0, a1, b0[4], b1[4];
    LOADA(a0, 0);
#pragma unroll
    for (int j = 0; j < 4; ++j) LOADB(b0[j], j, 0);

    for (int k0 = 0; k0 < K; k0 += 64) {
        // prefetch second half of this 64-wide block
        LOADA(a1, k0 + 32);
#pragma unroll
        for (int j = 0; j < 4; ++j) LOADB(b1[j], j, k0 + 32);
        // compute on first half while second half is in flight
#pragma unroll
        for (int j = 0; j < 4; ++j)
            acc[j] = __builtin_amdgcn_wmma_f32_16x16x32_bf16(
                false, a0.v, false, b0[j].v, (short)0, acc[j], false, false);
        // prefetch first half of next block
        if (k0 + 64 < K) {
            LOADA(a0, k0 + 64);
#pragma unroll
            for (int j = 0; j < 4; ++j) LOADB(b0[j], j, k0 + 64);
        }
#pragma unroll
        for (int j = 0; j < 4; ++j)
            acc[j] = __builtin_amdgcn_wmma_f32_16x16x32_bf16(
                false, a1.v, false, b1[j].v, (short)0, acc[j], false, false);
    }
#undef LOADA
#undef LOADB

    // epilogue: C/D layout -> VGPR r: row = m0 + half*8 + r, col = n0 + j*16 + sub
#pragma unroll
    for (int j = 0; j < 4; ++j) {
        int col = n0 + j * 16 + sub;
        if (col >= N) continue;
        float bv = bias ? bias[col] : 0.f;
#pragma unroll
        for (int r = 0; r < 8; ++r) {
            int row = m0 + half * 8 + r;
            if (row >= M) continue;
            float v = acc[j][r] + bv;
            size_t ci = (size_t)row * ldc + col;
            if (flags & 1) v += C[ci];
            if (flags & 2) v = sigf(v);
            if (flags & 4) v = (dec_len[row] > t) ? v : 0.f;
            C[ci] = v;
        }
    }
}

// ---------------- attention scores: (relu(att1+att2) @ wf) + bf ----------------
__global__ void k_scores(const float* __restrict__ att1, const float* __restrict__ att2,
                         const float* __restrict__ wf, const float* __restrict__ bfp,
                         float* __restrict__ scores) {
    int wid = blockIdx.x * (blockDim.x >> 5) + (threadIdx.x >> 5);
    int lane = threadIdx.x & 31;
    if (wid >= B_ * P_) return;
    int b = wid / P_, p = wid - b * P_;
    const float* a1 = att1 + ((size_t)b * P_ + p) * A_;
    const float* a2 = att2 + (size_t)b * A_;
    float s = 0.f;
    for (int a = lane; a < A_; a += 32) {
        float v = a1[a] + a2[a];
        v = v > 0.f ? v : 0.f;
        s += v * wf[a];
    }
    for (int o = 16; o; o >>= 1) s += __shfl_down(s, o);
    if (lane == 0) scores[wid] = s + bfp[0];
}

__global__ void k_softmax(const float* __restrict__ scores, float* __restrict__ alpha,
                          float* __restrict__ out_alphas, const int* __restrict__ dec_len, int t) {
    int b = blockIdx.x, tid = threadIdx.x;
    __shared__ float sh[256];
    float m = -1e30f;
    for (int p = tid; p < P_; p += 256) m = fmaxf(m, scores[b * P_ + p]);
    sh[tid] = m; __syncthreads();
    for (int o = 128; o; o >>= 1) { if (tid < o) sh[tid] = fmaxf(sh[tid], sh[tid + o]); __syncthreads(); }
    float mx = sh[0]; __syncthreads();
    float s = 0.f;
    for (int p = tid; p < P_; p += 256) { float e = __expf(scores[b * P_ + p] - mx); alpha[b * P_ + p] = e; s += e; }
    sh[tid] = s; __syncthreads();
    for (int o = 128; o; o >>= 1) { if (tid < o) sh[tid] += sh[tid + o]; __syncthreads(); }
    float inv = 1.f / sh[0];
    float msk = (dec_len[b] > t) ? 1.f : 0.f;
    for (int p = tid; p < P_; p += 256) {
        float a = alpha[b * P_ + p] * inv;
        alpha[b * P_ + p] = a;                                    // unmasked for awe
        out_alphas[((size_t)b * T_ + t) * P_ + p] = a * msk;      // masked output
    }
}

__global__ void k_awe(const float* __restrict__ alpha, const unsigned short* __restrict__ enc_bf,
                      float* __restrict__ awe) {
    int tid = blockIdx.x * blockDim.x + threadIdx.x;   // B*D
    int b = tid / D_, d = tid - b * D_;
    const unsigned short* e = enc_bf + (size_t)b * P_ * D_ + d;
    const float* al = alpha + b * P_;
    float s = 0.f;
    for (int p = 0; p < P_; ++p) s += al[p] * bf2f(e[(size_t)p * D_]);
    awe[tid] = s;
}

__global__ void k_build_x(const int* __restrict__ caps_s, const float* __restrict__ emb_W,
                          const float* __restrict__ gate, const float* __restrict__ awe,
                          unsigned short* __restrict__ x_bf, int t) {
    int tid = blockIdx.x * blockDim.x + threadIdx.x;   // B*(E+D)
    int b = tid / (E_ + D_), q = tid - b * (E_ + D_);
    float v;
    if (q < E_) {
        int cap = caps_s[b * S_ + t];
        v = emb_W[(size_t)cap * E_ + q];
    } else {
        int d = q - E_;
        v = gate[b * D_ + d] * awe[b * D_ + d];
    }
    x_bf[tid] = f2bf(v);
}

__global__ void k_lstm(const float* __restrict__ g, float* __restrict__ h, float* __restrict__ c,
                       unsigned short* __restrict__ h_bf) {
    int tid = blockIdx.x * blockDim.x + threadIdx.x;   // B*H
    int b = tid / H_, j = tid - b * H_;
    const float* gr = g + (size_t)b * 4 * H_;
    float gi = gr[j], gf = gr[H_ + j], gg = gr[2 * H_ + j], go = gr[3 * H_ + j];
    float cn = sigf(gf) * c[tid] + sigf(gi) * tanhf(gg);
    float hn = sigf(go) * tanhf(cn);
    c[tid] = cn; h[tid] = hn; h_bf[tid] = f2bf(hn);
}

// ---------------- host ----------------
extern "C" void kernel_launch(void* const* d_in, const int* in_sizes, int n_in,
                              void* d_out, int out_size, void* d_ws, size_t ws_size,
                              hipStream_t stream) {
    (void)in_sizes; (void)n_in; (void)out_size; (void)ws_size;
    const float* features = (const float*)d_in[0];
    const int*   captions = (const int*)d_in[1];
    const int*   cap_len  = (const int*)d_in[2];
    const float* emb_W = (const float*)d_in[3];
    const float* We  = (const float*)d_in[4];  const float* be  = (const float*)d_in[5];
    const float* Wd  = (const float*)d_in[6];  const float* bd  = (const float*)d_in[7];
    const float* wf  = (const float*)d_in[8];  const float* bf  = (const float*)d_in[9];
    const float* Wih = (const float*)d_in[10]; const float* Whh = (const float*)d_in[11];
    const float* bih = (const float*)d_in[12]; const float* bhh = (const float*)d_in[13];
    const float* Wh0 = (const float*)d_in[14]; const float* bh0 = (const float*)d_in[15];
    const float* Wc0 = (const float*)d_in[16]; const float* bc0 = (const float*)d_in[17];
    const float* Wbeta = (const float*)d_in[18]; const float* bbeta = (const float*)d_in[19];
    const float* Wout  = (const float*)d_in[20]; const float* bout  = (const float*)d_in[21];
    float* out = (float*)d_out;

    // output layout (floats): preds | caps | lens | alphas | sort_ind
    const size_t OFF_PREDS = 0;
    const size_t OFF_CAPS  = (size_t)B_ * T_ * V_;
    const size_t OFF_LENS  = OFF_CAPS + (size_t)B_ * S_;
    const size_t OFF_ALPH  = OFF_LENS + B_;
    const size_t OFF_SORT  = OFF_ALPH + (size_t)B_ * T_ * P_;

    // workspace carve (256-B aligned)
    size_t off = 0;
    auto carve = [&](size_t bytes) -> void* {
        off = (off + 255) & ~(size_t)255;
        void* p = (char*)d_ws + off; off += bytes; return p;
    };
    int* sort_ind = (int*)carve(B_ * 4);
    int* lens_s   = (int*)carve(B_ * 4);
    int* dec_len  = (int*)carve(B_ * 4);
    int* caps_s   = (int*)carve(B_ * S_ * 4);
    unsigned short* enc_bf  = (unsigned short*)carve((size_t)B_ * P_ * D_ * 2);
    unsigned short* mean_bf = (unsigned short*)carve((size_t)B_ * D_ * 2);
    unsigned short* WdT     = (unsigned short*)carve((size_t)A_ * H_ * 2);       // [A,H]
    unsigned short* WeT     = (unsigned short*)carve((size_t)A_ * D_ * 2);       // [A,D]
    unsigned short* WbetaT  = (unsigned short*)carve((size_t)D_ * H_ * 2);       // [D,H]
    unsigned short* WoutT   = (unsigned short*)carve((size_t)V_ * H_ * 2);       // [V,H]
    unsigned short* Wh0T    = (unsigned short*)carve((size_t)H_ * D_ * 2);       // [H,D]
    unsigned short* Wc0T    = (unsigned short*)carve((size_t)H_ * D_ * 2);
    unsigned short* WihB    = (unsigned short*)carve((size_t)4 * H_ * (E_ + D_) * 2); // [4H,E+D] == B^T
    unsigned short* WhhB    = (unsigned short*)carve((size_t)4 * H_ * H_ * 2);        // [4H,H]  == B^T
    float* bsum  = (float*)carve(4 * H_ * 4);
    float* att1  = (float*)carve((size_t)B_ * P_ * A_ * 4);
    float* h     = (float*)carve((size_t)B_ * H_ * 4);
    float* c     = (float*)carve((size_t)B_ * H_ * 4);
    unsigned short* h_bf = (unsigned short*)carve((size_t)B_ * H_ * 2);
    float* att2  = (float*)carve((size_t)B_ * A_ * 4);
    float* scores= (float*)carve((size_t)B_ * P_ * 4);
    float* alpha = (float*)carve((size_t)B_ * P_ * 4);
    float* awe   = (float*)carve((size_t)B_ * D_ * 4);
    float* gate  = (float*)carve((size_t)B_ * D_ * 4);
    unsigned short* x_bf = (unsigned short*)carve((size_t)B_ * (E_ + D_) * 2);
    float* gbuf  = (float*)carve((size_t)B_ * 4 * H_ * 4);

    auto gemm = [&](const unsigned short* A, int lda, long long sA,
                    const unsigned short* Bt, int ldb,
                    float* C, int ldc, long long sC, const float* bias,
                    int M, int N, int K, int flags, const int* dl, int t, int batch) {
        dim3 g((M + 15) / 16, (N + 63) / 64, batch);
        k_gemm<<<g, 32, 0, stream>>>(A, lda, sA, Bt, ldb, C, ldc, sC, bias, M, N, K, flags, dl, t);
    };

    // ---- setup ----
    k_sort<<<1, 64, 0, stream>>>(cap_len, captions, sort_ind, lens_s, dec_len, caps_s,
                                 out + OFF_CAPS, out + OFF_LENS, out + OFF_SORT);
    k_gather_enc<<<(unsigned)(((size_t)B_ * P_ * D_) / 256), 256, 0, stream>>>(features, sort_ind, enc_bf);
    k_mean<<<(B_ * D_) / 256, 256, 0, stream>>>(enc_bf, mean_bf);

    auto convT = [&](const float* W, unsigned short* Wt, int K, int N) {
        long long n = (long long)K * N;
        k_convT<<<(unsigned)((n + 255) / 256), 256, 0, stream>>>(W, Wt, K, N);
    };
    convT(Wd, WdT, H_, A_);
    convT(We, WeT, D_, A_);
    convT(Wbeta, WbetaT, H_, D_);
    convT(Wout, WoutT, H_, V_);
    convT(Wh0, Wh0T, D_, H_);
    convT(Wc0, Wc0T, D_, H_);
    k_conv<<<(unsigned)(((long long)4 * H_ * (E_ + D_) + 255) / 256), 256, 0, stream>>>(Wih, WihB, (long long)4 * H_ * (E_ + D_));
    k_conv<<<(unsigned)(((long long)4 * H_ * H_ + 255) / 256), 256, 0, stream>>>(Whh, WhhB, (long long)4 * H_ * H_);
    k_bsum<<<(4 * H_) / 256, 256, 0, stream>>>(bih, bhh, bsum, 4 * H_);

    // h0, c0
    gemm(mean_bf, D_, 0, Wh0T, D_, h, H_, 0, bh0, B_, H_, D_, 0, nullptr, 0, 1);
    gemm(mean_bf, D_, 0, Wc0T, D_, c, H_, 0, bc0, B_, H_, D_, 0, nullptr, 0, 1);
    k_conv<<<(B_ * H_) / 256, 256, 0, stream>>>(h, h_bf, (long long)B_ * H_);

    // att1[b] = enc[b] @ We + be  (batched over B)
    gemm(enc_bf, D_, (long long)P_ * D_, WeT, D_, att1, A_, (long long)P_ * A_, be,
         P_, A_, D_, 0, nullptr, 0, B_);

    // ---- decode steps ----
    for (int t = 0; t < T_; ++t) {
        // att2 = h @ Wd + bd
        gemm(h_bf, H_, 0, WdT, H_, att2, A_, 0, bd, B_, A_, H_, 0, nullptr, 0, 1);
        // scores + softmax -> alpha (and masked alphas out)
        k_scores<<<(B_ * P_ + 7) / 8, 256, 0, stream>>>(att1, att2, wf, bf, scores);
        k_softmax<<<B_, 256, 0, stream>>>(scores, alpha, out + OFF_ALPH, dec_len, t);
        // awe = sum_p alpha * enc
        k_awe<<<(B_ * D_) / 256, 256, 0, stream>>>(alpha, enc_bf, awe);
        // gate = sigmoid(h @ Wbeta + bbeta)
        gemm(h_bf, H_, 0, WbetaT, H_, gate, D_, 0, bbeta, B_, D_, H_, 2, nullptr, 0, 1);
        // x = [emb_t, gate*awe] (bf16)
        k_build_x<<<(B_ * (E_ + D_)) / 256, 256, 0, stream>>>(caps_s, emb_W, gate, awe, x_bf, t);
        // g = x @ Wih^T + (bih+bhh); g += h @ Whh^T
        gemm(x_bf, E_ + D_, 0, WihB, E_ + D_, gbuf, 4 * H_, 0, bsum, B_, 4 * H_, E_ + D_, 0, nullptr, 0, 1);
        gemm(h_bf, H_, 0, WhhB, H_, gbuf, 4 * H_, 0, nullptr, B_, 4 * H_, H_, 1, nullptr, 0, 1);
        // LSTM pointwise -> h, c, h_bf
        k_lstm<<<(B_ * H_) / 256, 256, 0, stream>>>(gbuf, h, c, h_bf);
        // preds[:, t, :] = mask * (h @ Wout + bout), written strided into d_out
        gemm(h_bf, H_, 0, WoutT, H_, out + OFF_PREDS + (size_t)t * V_, T_ * V_, 0, bout,
             B_, V_, H_, 4, dec_len, t, 1);
    }
}
